// WeightOnlyInt4Linear_82291573391877
// MI455X (gfx1250) — matrix operand order repaired and checked by reading, not compiled, not run on hardware
//
#include <hip/hip_runtime.h>

// WeightOnlyInt4Linear: y[m,n] = sum_k x[m,k] * dequant(W)[n,k] + bias[n]
// M = 16384, N = 4096, K = 4096.  f16 WMMA, f32 accumulate, scale+bias epilogue.
// Block tile 128(M) x 256(N), BK=64; 8 waves, each computing 64x64.
// Pipeline: global_prefetch (+2 tiles) -> register prefetch (+1 tile) -> LDS -> WMMA.

typedef _Float16 f16;
typedef __attribute__((ext_vector_type(2)))  _Float16     h2;
typedef __attribute__((ext_vector_type(16))) _Float16     v16h;
typedef __attribute__((ext_vector_type(8)))  float        v8f;
typedef __attribute__((ext_vector_type(4)))  unsigned int u32x4;
typedef __attribute__((ext_vector_type(4)))  float        f32x4;

#define IN_F   4096
#define OUT_F  4096
#define MROWS  16384
#define BM 128
#define BN 256
#define BK 64
#define LDA 72   // BK + 8 halfs padding: 144B row stride, 16B aligned, conflict-free
#define LDB 72

// ---- speculative L2 prefetch, two tiles ahead (global_prefetch_b8) ----
__device__ __forceinline__ void prefetch_tile(const float* __restrict__ x,
                                              const signed char* __restrict__ wq,
                                              int block_m, int block_n, int k0, int tid) {
  #pragma unroll
  for (int i = 0; i < 4; ++i) {
    int idx = tid + i * 256;
    int row = idx >> 3;
    int c8  = (idx & 7) << 3;
    __builtin_prefetch(x + (size_t)(block_m + row) * IN_F + k0 + c8, 0, 0);
  }
  // one line per W row (rows have 2KB stride; 32B/tile/row)
  __builtin_prefetch(wq + (size_t)(block_n + tid) * (IN_F / 2) + (k0 >> 1), 0, 0);
}

// ---- raw tile prefetch into registers (global latency hidden behind WMMA) ----
__device__ __forceinline__ void load_tile(const float* __restrict__ x,
                                          const signed char* __restrict__ wq,
                                          int block_m, int block_n, int k0, int tid,
                                          f32x4 ar[8], u32x4 br[2]) {
  #pragma unroll
  for (int i = 0; i < 4; ++i) {
    int idx = tid + i * 256;            // 1024 float8-chunks: 128 rows x 8
    int row = idx >> 3;
    int c8  = (idx & 7) << 3;
    const float* p = x + (size_t)(block_m + row) * IN_F + k0 + c8;
    ar[2*i]   = __builtin_nontemporal_load((const f32x4*)p);      // streamed once
    ar[2*i+1] = __builtin_nontemporal_load((const f32x4*)p + 1);
  }
  // B: 256 rows x 32 packed bytes = 8KB -> 32B per thread (2 x 16B chunks)
  #pragma unroll
  for (int t = 0; t < 2; ++t) {
    int row  = (tid >> 1) + t * 128;
    int part = tid & 1;
    br[t] = *(const u32x4*)(wq + (size_t)(block_n + row) * (IN_F / 2) + (k0 >> 1) + part * 16);
  }
}

// ---- convert + stage to LDS ----
__device__ __forceinline__ void stage_tile(unsigned short* lds_a, unsigned short* lds_b,
                                           int tid, const f32x4 ar[8], const u32x4 br[2]) {
  // A: fp32 -> f16 with v_cvt_pk_rtz_f16_f32, b128 stores
  #pragma unroll
  for (int i = 0; i < 4; ++i) {
    int idx = tid + i * 256;
    int row = idx >> 3;
    int c8  = (idx & 7) << 3;
    u32x4 pk;
    pk[0] = __builtin_bit_cast(unsigned int, __builtin_amdgcn_cvt_pkrtz(ar[2*i][0],   ar[2*i][1]));
    pk[1] = __builtin_bit_cast(unsigned int, __builtin_amdgcn_cvt_pkrtz(ar[2*i][2],   ar[2*i][3]));
    pk[2] = __builtin_bit_cast(unsigned int, __builtin_amdgcn_cvt_pkrtz(ar[2*i+1][0], ar[2*i+1][1]));
    pk[3] = __builtin_bit_cast(unsigned int, __builtin_amdgcn_cvt_pkrtz(ar[2*i+1][2], ar[2*i+1][3]));
    *(u32x4*)&lds_a[row * LDA + c8] = pk;
  }
  // B: int4 -> f16 via magic number: (0x6400 | (nib^8)) is f16(1024 + s + 8);
  // packed add of -1032.0h yields the exact signed int4 value.
  const h2 kMagicNeg = __builtin_bit_cast(h2, 0xE408E408u);
  #pragma unroll
  for (int t = 0; t < 2; ++t) {
    int row  = (tid >> 1) + t * 128;
    int part = tid & 1;
    unsigned short* dst = &lds_b[row * LDB + part * 32];
    #pragma unroll
    for (int d = 0; d < 4; ++d) {
      unsigned int v  = br[t][d];
      unsigned int lo = ( v       & 0x0F0F0F0Fu) ^ 0x08080808u;  // even k, biased +8
      unsigned int hi = ((v >> 4) & 0x0F0F0F0Fu) ^ 0x08080808u;  // odd  k, biased +8
      u32x4 o;
      #pragma unroll
      for (int j = 0; j < 4; ++j) {
        unsigned int sel = 0x0C000C00u | (unsigned)j | ((unsigned)(4 + j) << 16);
        unsigned int tt  = __builtin_amdgcn_perm(hi, lo, sel) | 0x64006400u;
        h2 f = __builtin_bit_cast(h2, tt) + kMagicNeg;            // v_pk_add_f16
        o[j] = __builtin_bit_cast(unsigned int, f);
      }
      *(u32x4*)&dst[d * 8] = o;
    }
  }
}

__global__ __launch_bounds__(256) void wo_int4_wmma(
    const float* __restrict__ x,
    const signed char* __restrict__ wq,
    const float* __restrict__ scales,
    const float* __restrict__ bias,
    float* __restrict__ out)
{
  __shared__ unsigned short lds_a[BM * LDA];   // 18.4 KB
  __shared__ unsigned short lds_b[BN * LDB];   // 36.9 KB

  const int tid  = threadIdx.x;
  const int lane = tid & 31;
  const int wv   = tid >> 5;
  const int wm   = wv >> 2;         // 0..1 (M): 64 rows each
  const int wn   = wv & 3;          // 0..3 (N): 64 cols each
  const int lrow = lane & 15;
  const int lhi  = lane >> 4;

  const int block_n = blockIdx.x * BN;
  const int block_m = blockIdx.y * BM;

  v8f acc[4][4] = {};               // wave: 64(M) x 64(N) = 16 wmma tiles

  f32x4 ar[8];
  u32x4 br[2];
  prefetch_tile(x, wq, block_m, block_n, BK, tid);
  load_tile(x, wq, block_m, block_n, 0, tid, ar, br);

  for (int k0 = 0; k0 < IN_F; k0 += BK) {
    stage_tile(lds_a, lds_b, tid, ar, br);
    __syncthreads();

    if (k0 + BK < IN_F) {           // register prefetch +1 tile; L2 prefetch +2 tiles
      load_tile(x, wq, block_m, block_n, k0 + BK, tid, ar, br);
      if (k0 + 2 * BK < IN_F)
        prefetch_tile(x, wq, block_m, block_n, k0 + 2 * BK, tid);
    }

    union Frag { v16h v; u32x4 q[2]; };
    #pragma unroll
    for (int kk = 0; kk < BK; kk += 32) {
      // preload fragments so ds_load_b128s pipeline ahead of the WMMA chain
      Frag af[4], bf[4];
      #pragma unroll
      for (int tm = 0; tm < 4; ++tm) {
        const unsigned short* p = &lds_a[(wm * 64 + tm * 16 + lrow) * LDA + kk + lhi * 8];
        af[tm].q[0] = *(const u32x4*)(p);
        af[tm].q[1] = *(const u32x4*)(p + 16);
      }
      #pragma unroll
      for (int tn = 0; tn < 4; ++tn) {
        const unsigned short* p = &lds_b[(wn * 64 + tn * 16 + lrow) * LDB + kk + lhi * 8];
        bf[tn].q[0] = *(const u32x4*)(p);
        bf[tn].q[1] = *(const u32x4*)(p + 16);
      }
      #pragma unroll
      for (int tn = 0; tn < 4; ++tn) {
        #pragma unroll
        for (int tm = 0; tm < 4; ++tm) {
          acc[tm][tn] = __builtin_amdgcn_wmma_f32_16x16x32_f16(
              false, af[tm].v, false, bf[tn].v, (short)0, acc[tm][tn], false, false);
        }
      }
    }
    __syncthreads();
  }

  // epilogue: scale + bias, non-temporal streaming stores (keep W resident in L2)
  #pragma unroll
  for (int tn = 0; tn < 4; ++tn) {
    const int n  = block_n + wn * 64 + tn * 16 + lrow;
    const float s  = scales[n];
    const float bb = bias[n];
    #pragma unroll
    for (int tm = 0; tm < 4; ++tm) {
      const int mbase = block_m + wm * 64 + tm * 16 + lhi * 8;
      #pragma unroll
      for (int r = 0; r < 8; ++r) {
        __builtin_nontemporal_store(acc[tm][tn][r] * s + bb,
                                    &out[(size_t)(mbase + r) * OUT_F + n]);
      }
    }
  }
}

extern "C" void kernel_launch(void* const* d_in, const int* in_sizes, int n_in,
                              void* d_out, int out_size, void* d_ws, size_t ws_size,
                              hipStream_t stream) {
  (void)in_sizes; (void)n_in; (void)out_size; (void)d_ws; (void)ws_size;
  const float*       xx = (const float*)d_in[0];
  const signed char* wq = (const signed char*)d_in[1];
  const float*       sc = (const float*)d_in[2];
  const float*       bi = (const float*)d_in[3];
  float*             yy = (float*)d_out;

  dim3 grid(OUT_F / BN, MROWS / BM);   // (16, 128)
  dim3 block(256);
  wo_int4_wmma<<<grid, block, 0, stream>>>(xx, wq, sc, bi, yy);
}
